// CrossOp_48095043780799
// MI455X (gfx1250) — matrix operand order repaired and checked by reading, not compiled, not run on hardware
//
#include <hip/hip_runtime.h>
#include <hip/hip_bf16.h>

typedef __attribute__((ext_vector_type(16))) __bf16 v16bf;
typedef __attribute__((ext_vector_type(8)))  float  v8f;

#define CIN   32
#define CO    32
#define HH    128
#define WD    128
#define KCH   9            // 3x3 taps, one K=32 chunk each (k index == ci)
#define ROWS  4            // output rows per workgroup
#define LROWS (ROWS + 2)   // with halo
#define LX    66           // 64-wide tile + 2 halo
#define LDS_ELEMS (LROWS * LX * CIN)      // 12672 per plane
#define FRAG_ELEMS (2 * KCH * 32 * 16)    // 9216 bf16 per (set,plane)

// ---------------------------------------------------------------------------
// Pre-swizzle weights into the WMMA A-operand layout (16-bit A 16x32, wave32):
// lane = half*16 + m ; element j: K = half*16 + j (contiguous k order used here, same as B; K == ci)
// Layout: [set(X,Y)][plane(hi,lo)][mtile][chunk][lane][j], bf16 split pairs.
// ---------------------------------------------------------------------------
__global__ void prep_frags(const float* __restrict__ Wt, __bf16* __restrict__ frags) {
    int idx = blockIdx.x * blockDim.x + threadIdx.x;
    if (idx >= 4 * FRAG_ELEMS) return;
    int which = idx / FRAG_ELEMS;          // 0:XHi 1:XLo 2:YHi 3:YLo
    int set   = which >> 1;
    int plane = which & 1;
    int rem   = idx % FRAG_ELEMS;
    int mtile = rem / (KCH * 32 * 16);
    int chunk = (rem / (32 * 16)) % KCH;
    int lane  = (rem / 16) % 32;
    int j     = rem & 15;
    int m     = lane & 15;
    int half  = lane >> 4;
    int ci    = half * 16 + j;                         // K index within chunk
    int co    = mtile * 16 + m;
    float  v  = Wt[(co * 64 + set * 32 + ci) * 9 + chunk];   // W: (32,64,3,3)
    __bf16 hi = (__bf16)v;
    __bf16 lo = (__bf16)(v - (float)hi);
    frags[idx] = plane ? lo : hi;
}

// ---------------------------------------------------------------------------
// Implicit-GEMM 3x3 conv, split-bf16 (hi/lo) for fp32-grade accuracy.
// Grid: (W/64, H/ROWS, n_img). 256 threads = 8 waves; wave -> (mtile, row g).
// Each wave: 4 N-tiles of 16 px on its row; 9 chunks x 3 WMMAs per tile.
// ---------------------------------------------------------------------------
__global__ __launch_bounds__(256)
void conv_wmma(const float* __restrict__ src,
               const __bf16* __restrict__ fragHi,
               const __bf16* __restrict__ fragLo,
               const float* __restrict__ addsrc,   // cx buffer or null
               const float* __restrict__ bias,     // bias or null
               float* __restrict__ out) {
    __shared__ __bf16 sHi[LDS_ELEMS];
    __shared__ __bf16 sLo[LDS_ELEMS];

    const int img = blockIdx.z;
    const int h0  = blockIdx.y * ROWS;
    const int w0  = blockIdx.x * 64;
    const int tid = threadIdx.x;
    const float* imgp = src + (size_t)img * (CIN * HH * WD);

    // Stage halo tile into LDS, pre-split into bf16 hi/lo planes.
    for (int idx = tid; idx < LDS_ELEMS; idx += 256) {
        int ci = idx / (LROWS * LX);
        int rr = idx % (LROWS * LX);
        int r  = rr / LX;
        int x  = rr % LX;
        int w  = w0 - 1 + x;
        int h  = h0 - 1 + r;
        float v = 0.0f;
        if ((unsigned)w < (unsigned)WD && (unsigned)h < (unsigned)HH)
            v = imgp[ci * (HH * WD) + h * WD + w];
        __bf16 hi = (__bf16)v;
        __bf16 lo = (__bf16)(v - (float)hi);
        int a = (r * LX + x) * CIN + ci;                 // [row][x][ci]
        sHi[a] = hi;
        sLo[a] = lo;
    }
    __syncthreads();

    const int wave  = tid >> 5;
    const int lane  = tid & 31;
    const int mtile = wave & 1;      // co 0-15 / 16-31
    const int g     = wave >> 1;     // output row within group (0..3)
    const int n     = lane & 15;     // N column
    const int half  = lane >> 4;

    v8f acc[4] = {};

    for (int chunk = 0; chunk < KCH; ++chunk) {
        int kh = chunk / 3;
        int kw = chunk % 3;
        const int foff = ((mtile * KCH + chunk) * 32 + lane) * 16;
        v16bf aHi = *(const v16bf*)(fragHi + foff);   // coalesced 32B/lane
        v16bf aLo = *(const v16bf*)(fragLo + foff);
        const int rl = g + kh;                        // LDS row (halo offset)
        #pragma unroll
        for (int t = 0; t < 4; ++t) {
            int x    = t * 16 + n + kw;               // <= 65
            int boff = (rl * LX + x) * CIN + half * 16;
            v16bf bHi = *(const v16bf*)(sHi + boff);  // ds_load_b128 x2
            v16bf bLo = *(const v16bf*)(sLo + boff);
            acc[t] = __builtin_amdgcn_wmma_f32_16x16x32_bf16(
                         false, aHi, false, bHi, (short)0, acc[t], false, false);
            acc[t] = __builtin_amdgcn_wmma_f32_16x16x32_bf16(
                         false, aLo, false, bHi, (short)0, acc[t], false, false);
            acc[t] = __builtin_amdgcn_wmma_f32_16x16x32_bf16(
                         false, aHi, false, bLo, (short)0, acc[t], false, false);
        }
    }

    // Epilogue: C/D layout -> VGPR r: M = half*8 + r, N = lane%16.
    const int hrow = h0 + g;
    #pragma unroll
    for (int t = 0; t < 4; ++t) {
        int w = w0 + t * 16 + n;
        #pragma unroll
        for (int r = 0; r < 8; ++r) {
            int co = mtile * 16 + half * 8 + r;
            float v = acc[t][r];
            if (bias)   v += bias[co];
            if (addsrc) v += addsrc[(size_t)((img >> 5) * CO + co) * (HH * WD) + hrow * WD + w];
            out[(size_t)(img * CO + co) * (HH * WD) + hrow * WD + w] = v;
        }
    }
}

// ---------------------------------------------------------------------------
// new_target[b,c,p] = mean over sy of interaction[b,sy,c,p]
// ---------------------------------------------------------------------------
__global__ void reduce_mean(const float* __restrict__ inter, float* __restrict__ outp) {
    int idx = blockIdx.x * blockDim.x + threadIdx.x;
    if (idx >= 4 * CO * HH * WD) return;
    int b   = idx >> 19;           // 32*128*128 = 524288 per batch
    int rem = idx & 524287;        // c*16384 + p
    float s = 0.0f;
    for (int sy = 0; sy < 32; ++sy)
        s += inter[(size_t)(b * 32 + sy) * 524288 + rem];
    outp[idx] = s * (1.0f / 32.0f);
}

extern "C" void kernel_launch(void* const* d_in, const int* in_sizes, int n_in,
                              void* d_out, int out_size, void* d_ws, size_t ws_size,
                              hipStream_t stream) {
    const float* target  = (const float*)d_in[0];   // (4,1,32,128,128)
    const float* support = (const float*)d_in[1];   // (4,32,32,128,128)
    const float* Wt      = (const float*)d_in[2];   // (32,64,3,3)
    const float* bias    = (const float*)d_in[3];   // (32,)

    float* new_target  = (float*)d_out;             // 2,097,152 floats
    float* interaction = (float*)d_out + 2097152;   // 67,108,864 floats

    float*  cxbuf = (float*)d_ws;                                  // 8 MiB
    __bf16* frags = (__bf16*)((char*)d_ws + (size_t)2097152 * 4);  // 72 KiB
    __bf16* fxHi = frags;
    __bf16* fxLo = frags + FRAG_ELEMS;
    __bf16* fyHi = frags + 2 * FRAG_ELEMS;
    __bf16* fyLo = frags + 3 * FRAG_ELEMS;

    // 1) Swizzle + split weights into WMMA A-fragment layout.
    prep_frags<<<(4 * FRAG_ELEMS + 255) / 256, 256, 0, stream>>>(Wt, frags);

    // 2) cx = conv(target, Wx) -> scratch  (4 images)
    conv_wmma<<<dim3(WD / 64, HH / ROWS, 4), 256, 0, stream>>>(
        target, fxHi, fxLo, nullptr, nullptr, cxbuf);

    // 3) interaction = conv(support, Wy) + cx + bias  (128 images)
    conv_wmma<<<dim3(WD / 64, HH / ROWS, 128), 256, 0, stream>>>(
        support, fyHi, fyLo, cxbuf, bias, interaction);

    // 4) new_target = mean over Sy
    reduce_mean<<<(4 * CO * HH * WD + 255) / 256, 256, 0, stream>>>(
        interaction, new_target);
}
